// SPConvFuse_2877628088615
// MI455X (gfx1250) — matrix operand order repaired and checked by reading, not compiled, not run on hardware
//
#include <hip/hip_runtime.h>
#include <hip/hip_bf16.h>
#include <stdint.h>

#define CIN  64
#define COUT 128
#define EPS  0.001f
#define WPACK (27 * 2 * 8 * 32 * 8)   // packed weight words per plane

typedef __attribute__((ext_vector_type(16))) __bf16       v16bf;
typedef __attribute__((ext_vector_type(8)))  float        v8f;
typedef __attribute__((ext_vector_type(8)))  unsigned int v8u;

__device__ inline unsigned int bf16_rne(float x) {
    unsigned int u = __float_as_uint(x);
    unsigned int r = u + 0x7fffu + ((u >> 16) & 1u);
    return r >> 16;
}
__device__ inline float bf16_to_f(unsigned int b) { return __uint_as_float(b << 16); }

// ---------------------------------------------------------------------------
// Pre-swizzle weights into WMMA B-operand layout (hi/lo bf16 planes, RNE).
// B (32x16, bf16): lane l -> column n = l&15, half h = l>>4; VGPR0-3 hold K = 8h..8h+7,
// VGPR4-7 hold K = 16+8h..16+8h+7 (2 consecutive K per VGPR, low16 first).
// Storage: Whi[((t*2+c)*8+nt)*32*8 + lane*8 + j]
// ---------------------------------------------------------------------------
__global__ void wprep_kernel(const float* __restrict__ W,
                             unsigned int* __restrict__ Whi,
                             unsigned int* __restrict__ Wlo) {
    int tid = blockIdx.x * blockDim.x + threadIdx.x;
    if (tid >= WPACK) return;
    int j  = tid & 7;
    int l  = (tid >> 3) & 31;
    int nt = (tid >> 8) & 7;
    int c  = (tid >> 11) & 1;
    int t  = tid >> 12;
    int n  = nt * 16 + (l & 15);
    int h  = l >> 4;
    int k  = c * 32 + 16 * (j >> 2) + 8 * h + 2 * (j & 3);
    float w0 = W[(t * CIN + k) * COUT + n];
    float w1 = W[(t * CIN + k + 1) * COUT + n];
    unsigned int h0 = bf16_rne(w0), h1 = bf16_rne(w1);
    Whi[tid] = h0 | (h1 << 16);
    unsigned int l0 = bf16_rne(w0 - bf16_to_f(h0));
    unsigned int l1 = bf16_rne(w1 - bf16_to_f(h1));
    Wlo[tid] = l0 | (l1 << 16);
}

// ---------------------------------------------------------------------------
// Build A-operand (16x32 bf16 per K-chunk) for one row, hi/lo split.
// Truncation split: hi = top 16 bits of fp32 (residual a-hi is EXACT in fp32),
// lo = top 16 bits of residual. Dropped error ~2^-16 relative — same order as
// the omitted lo*lo WMMA term. Much cheaper than RNE (no add3/bfe chains).
// A layout: lane l -> row m = l&15, h = l>>4; chunk c: VGPR0-3 hold K=32c+8h..+7,
// VGPR4-7 hold K=32c+16+8h..+7 (2 consecutive K per VGPR, low16 first).
// ---------------------------------------------------------------------------
__device__ inline void build_A(const float* rowptr, bool valid, int h,
                               v16bf Ahi[2], v16bf Alo[2]) {
    float vals[32];
    if (valid) {
        const float4* p = (const float4*)rowptr;
#pragma unroll
        for (int g = 0; g < 4; ++g) {          // groups: K base = (g&1)*16 + (g>>1)*32 + 8h
            float4 v0 = p[g * 4 + 2 * h];
            float4 v1 = p[g * 4 + 2 * h + 1];
            vals[g*8+0]=v0.x; vals[g*8+1]=v0.y; vals[g*8+2]=v0.z; vals[g*8+3]=v0.w;
            vals[g*8+4]=v1.x; vals[g*8+5]=v1.y; vals[g*8+6]=v1.z; vals[g*8+7]=v1.w;
        }
    } else {
#pragma unroll
        for (int i = 0; i < 32; ++i) vals[i] = 0.0f;
    }
#pragma unroll
    for (int c = 0; c < 2; ++c) {
        v8u wh, wl;
#pragma unroll
        for (int j = 0; j < 8; ++j) {
            int g = c * 2 + (j >> 2);
            int e = (j & 3) * 2;
            unsigned int u0 = __float_as_uint(vals[g * 8 + e]);
            unsigned int u1 = __float_as_uint(vals[g * 8 + e + 1]);
            unsigned int m0 = u0 & 0xffff0000u;        // hi as fp32 bits
            unsigned int m1 = u1 & 0xffff0000u;
            wh[j] = (m0 >> 16) | m1;                   // pack(hi0, hi1)
            float r0 = vals[g * 8 + e]     - __uint_as_float(m0);   // exact
            float r1 = vals[g * 8 + e + 1] - __uint_as_float(m1);   // exact
            wl[j] = (__float_as_uint(r0) >> 16) | (__float_as_uint(r1) & 0xffff0000u);
        }
        Ahi[c] = __builtin_bit_cast(v16bf, wh);
        Alo[c] = __builtin_bit_cast(v16bf, wl);
    }
}

__device__ inline v16bf load_B(const unsigned int* base) {
    const uint4* p = (const uint4*)base;
    uint4 b0 = p[0], b1 = p[1];
    v8u w;
    w[0]=b0.x; w[1]=b0.y; w[2]=b0.z; w[3]=b0.w;
    w[4]=b1.x; w[5]=b1.y; w[6]=b1.z; w[7]=b1.w;
    return __builtin_bit_cast(v16bf, w);
}

// bf16x3: acc += A*B with near-fp32 accuracy (drop lo*lo term)
#define WMMA3(Ah, Al, Bh, Bl, acc)                                                              \
    acc = __builtin_amdgcn_wmma_f32_16x16x32_bf16(false, Al, false, Bh, (short)0, acc, false, false); \
    acc = __builtin_amdgcn_wmma_f32_16x16x32_bf16(false, Ah, false, Bl, (short)0, acc, false, false); \
    acc = __builtin_amdgcn_wmma_f32_16x16x32_bf16(false, Ah, false, Bh, (short)0, acc, false, false);

// ---------------------------------------------------------------------------
// Center tap: out[r,:] = feats[r,:] @ W[13] + bias.
// One wave per 32-row super-tile (two 16-row WMMA tiles sharing B loads).
// ---------------------------------------------------------------------------
__global__ void __launch_bounds__(256) center_kernel(
        const float* __restrict__ f1, const float* __restrict__ f2,
        const float* __restrict__ bias,
        const unsigned int* __restrict__ Whi, const unsigned int* __restrict__ Wlo,
        float* __restrict__ out, int N1, int N) {
    int wave = blockIdx.x * 8 + (threadIdx.x >> 5);
    int row0 = wave * 32;
    if (row0 >= N) return;
    int lane = threadIdx.x & 31;
    int m = lane & 15, h = lane >> 4;

    v16bf Ahi0[2], Alo0[2], Ahi1[2], Alo1[2];
    {
        int r = row0 + m;
        const float* rp = (r < N1) ? (f1 + (size_t)r * CIN)
                                   : (f2 + (size_t)(r - N1) * CIN);
        build_A(rp, r < N, h, Ahi0, Alo0);
    }
    {
        int r = row0 + 16 + m;
        const float* rp = (r < N1) ? (f1 + (size_t)r * CIN)
                                   : (f2 + (size_t)(r - N1) * CIN);
        build_A(rp, r < N, h, Ahi1, Alo1);
    }
    bool full = (row0 + 32 <= N);          // uniform: unconditional stores on fast path

#pragma unroll
    for (int nt = 0; nt < 8; ++nt) {
        float b = bias[nt * 16 + m];
        v8f acc0, acc1;
#pragma unroll
        for (int j = 0; j < 8; ++j) { acc0[j] = b; acc1[j] = b; }
#pragma unroll
        for (int c = 0; c < 2; ++c) {
            size_t off = ((size_t)((13 * 2 + c) * 8 + nt) * 32 + lane) * 8;
            v16bf Bh = load_B(Whi + off);
            v16bf Bl = load_B(Wlo + off);
            WMMA3(Ahi0[c], Alo0[c], Bh, Bl, acc0);
            WMMA3(Ahi1[c], Alo1[c], Bh, Bl, acc1);
        }
        if (full) {
#pragma unroll
            for (int j = 0; j < 8; ++j) {
                int r0 = row0 + j + 8 * h;            // C/D layout: VGPR j -> M = j + 8h
                out[(size_t)r0 * COUT + nt * 16 + m] = acc0[j];
                out[(size_t)(r0 + 16) * COUT + nt * 16 + m] = acc1[j];
            }
        } else {
#pragma unroll
            for (int j = 0; j < 8; ++j) {
                int r0 = row0 + j + 8 * h;
                if (r0 < N)      out[(size_t)r0 * COUT + nt * 16 + m] = acc0[j];
                if (r0 + 16 < N) out[(size_t)(r0 + 16) * COUT + nt * 16 + m] = acc1[j];
            }
        }
    }
}

// ---------------------------------------------------------------------------
// 26 offset taps: gather 16 rulebook rows, GEMM, atomic scatter-add.
// Padding entries use index N (zero row / dropped output).
// ---------------------------------------------------------------------------
__global__ void __launch_bounds__(256) taps_kernel(
        const float* __restrict__ f1, const float* __restrict__ f2,
        const int* __restrict__ G, const int* __restrict__ Sc,
        const unsigned int* __restrict__ Whi, const unsigned int* __restrict__ Wlo,
        float* __restrict__ out, int N1, int N, int Pmax, int tilesPerTap) {
    int wave = blockIdx.x * 8 + (threadIdx.x >> 5);
    if (wave >= 26 * tilesPerTap) return;
    int tap = wave / tilesPerTap;
    int p0  = (wave - tap * tilesPerTap) * 16;
    // padding is a contiguous suffix: if first scatter idx is the dummy row, skip tile
    if (Sc[(size_t)tap * Pmax + p0] >= N) return;

    int lane = threadIdx.x & 31;
    int m = lane & 15, h = lane >> 4;
    int p = p0 + m;
    int g = (p < Pmax) ? G[(size_t)tap * Pmax + p] : N;
    bool valid = g < N;
    const float* rp = (g < N1) ? (f1 + (size_t)g * CIN)
                               : (f2 + (size_t)(g - N1) * CIN);
    v16bf Ahi[2], Alo[2];
    build_A(rp, valid, h, Ahi, Alo);

    int sidx[8];
#pragma unroll
    for (int j = 0; j < 8; ++j) {
        int pp = p0 + j + 8 * h;
        sidx[j] = (pp < Pmax) ? Sc[(size_t)tap * Pmax + pp] : N;
    }
    int t = tap + (tap >= 13 ? 1 : 0);                // skip center weight
#pragma unroll
    for (int nt = 0; nt < 8; ++nt) {
        v8f acc;
#pragma unroll
        for (int j = 0; j < 8; ++j) acc[j] = 0.0f;
#pragma unroll
        for (int c = 0; c < 2; ++c) {
            size_t off = ((size_t)((t * 2 + c) * 8 + nt) * 32 + lane) * 8;
            v16bf Bh = load_B(Whi + off);
            v16bf Bl = load_B(Wlo + off);
            WMMA3(Ahi[c], Alo[c], Bh, Bl, acc);
        }
#pragma unroll
        for (int j = 0; j < 8; ++j) {
            int s = sidx[j];
            if (s < N) atomicAdd(&out[(size_t)s * COUT + nt * 16 + m], acc[j]);
        }
    }
}

// ---------------------------------------------------------------------------
// BatchNorm over all N rows + ReLU
// ---------------------------------------------------------------------------
__global__ void zero_stats_kernel(float* stats) {
    if (threadIdx.x < 512) stats[threadIdx.x] = 0.0f;
}

__global__ void bn_stats_kernel(const float* __restrict__ out,
                                float* __restrict__ stats, int N) {
    int c = threadIdx.x;                               // 0..127
    float s = 0.f, sq = 0.f;
    for (int r = blockIdx.x; r < N; r += gridDim.x) {
        float v = out[(size_t)r * COUT + c];
        s += v; sq += v * v;
    }
    atomicAdd(&stats[c], s);
    atomicAdd(&stats[128 + c], sq);
}

__global__ void bn_scale_kernel(const float* __restrict__ gamma,
                                const float* __restrict__ beta,
                                float* stats, int N) {
    int c = threadIdx.x;
    float inv_n = 1.0f / (float)N;
    float mean = stats[c] * inv_n;
    float var  = stats[128 + c] * inv_n - mean * mean;
    float inv  = rsqrtf(var + EPS);
    float sc   = gamma[c] * inv;
    stats[256 + c] = sc;
    stats[384 + c] = beta[c] - mean * sc;
}

__global__ void bn_apply_kernel(float* __restrict__ out,
                                const float* __restrict__ stats, int total4) {
    int i = blockIdx.x * blockDim.x + threadIdx.x;
    if (i >= total4) return;
    float4* o4 = (float4*)out;
    float4 v = o4[i];
    int c0 = (i * 4) & 127;
    float4 sc = *(const float4*)(stats + 256 + c0);
    float4 sh = *(const float4*)(stats + 384 + c0);
    v.x = fmaxf(v.x * sc.x + sh.x, 0.f);
    v.y = fmaxf(v.y * sc.y + sh.y, 0.f);
    v.z = fmaxf(v.z * sc.z + sh.z, 0.f);
    v.w = fmaxf(v.w * sc.w + sh.w, 0.f);
    o4[i] = v;
}

// ---------------------------------------------------------------------------
extern "C" void kernel_launch(void* const* d_in, const int* in_sizes, int n_in,
                              void* d_out, int out_size, void* d_ws, size_t ws_size,
                              hipStream_t stream) {
    const float* f1    = (const float*)d_in[0];
    const float* f2    = (const float*)d_in[1];
    const float* W     = (const float*)d_in[2];
    const float* bias  = (const float*)d_in[3];
    const float* gamma = (const float*)d_in[4];
    const float* beta  = (const float*)d_in[5];
    const int*   G     = (const int*)d_in[6];
    const int*   Sc    = (const int*)d_in[7];

    int N1   = in_sizes[0] / CIN;
    int N    = N1 + in_sizes[1] / CIN;
    int Pmax = in_sizes[6] / 26;

    float* out = (float*)d_out;
    unsigned int* Whi = (unsigned int*)d_ws;
    unsigned int* Wlo = Whi + WPACK;
    float* stats = (float*)(Wlo + WPACK);              // 512 floats: sum|sumsq|scale|shift

    hipLaunchKernelGGL(zero_stats_kernel, dim3(1), dim3(512), 0, stream, stats);
    hipLaunchKernelGGL(wprep_kernel, dim3((WPACK + 255) / 256), dim3(256), 0, stream,
                       W, Whi, Wlo);

    int waves32 = (N + 31) / 32;
    hipLaunchKernelGGL(center_kernel, dim3((waves32 + 7) / 8), dim3(256), 0, stream,
                       f1, f2, bias, Whi, Wlo, out, N1, N);

    int tilesPerTap = (Pmax + 15) / 16;
    int waves = 26 * tilesPerTap;
    hipLaunchKernelGGL(taps_kernel, dim3((waves + 7) / 8), dim3(256), 0, stream,
                       f1, f2, G, Sc, Whi, Wlo, out, N1, N, Pmax, tilesPerTap);

    hipLaunchKernelGGL(bn_stats_kernel, dim3(2048), dim3(128), 0, stream, out, stats, N);
    hipLaunchKernelGGL(bn_scale_kernel, dim3(1), dim3(128), 0, stream, gamma, beta, stats, N);

    int total4 = (N * COUT) / 4;
    hipLaunchKernelGGL(bn_apply_kernel, dim3((total4 + 255) / 256), dim3(256), 0, stream,
                       out, stats, total4);
}
